// DARTSCell_19971597926404
// MI455X (gfx1250) — matrix operand order repaired and checked by reading, not compiled, not run on hardware
//
#include <hip/hip_runtime.h>

// ---------------------------------------------------------------------------
// DARTS RNN cell on gfx1250 (MI455X).
// One persistent kernel runs the whole 128-step scan; the 9 dependent GEMMs
// per step run on v_wmma_f32_16x16x32_bf16 (f32 accumulate, f32 gating),
// separated by device-scope spin barriers instead of kernel launches.
// Each wave owns a 64x16 output block (4 M-tiles x 1 N-tile) and both the
// c/h halves of the projection: 8 WMMA chains per wave, weights read only
// 4x per GEMM out of L2.  All hot-path accesses are forced to address
// space 1 so they lower to global_load/store (not flat).
// ---------------------------------------------------------------------------

typedef __attribute__((ext_vector_type(16))) __bf16 v16bf;
typedef __attribute__((ext_vector_type(8)))  __bf16 v8bf;
typedef __attribute__((ext_vector_type(8)))  float  v8f;

#define AS1 __attribute__((address_space(1)))
#define GCP(T, p) ((const AS1 T*)(unsigned long long)(p))   // global const ptr
#define GP(T, p)  ((AS1 T*)(unsigned long long)(p))         // global ptr

#define T_STEPS 128
#define BATCH   256
#define NHID    850
#define K0      1700    // ninp + nhid for step-0 GEMM
#define K0P     1728    // K0 padded to 32
#define KP      864     // nhid padded to 32
#define NWCOLS  1728    // 2*nhid padded (Wt column count)
#define BH      (BATCH * NHID)
#define NTILES  54      // ceil(850/16)
#define NBLK    27      // 216 waves = (256/64) * 54, 8 waves/block
#define NTHREAD (NBLK * 256)

// genotype, nibble-packed: pred[i] = (PREDS>>(4i))&15, act[i] = (ACTS>>(4i))&15
// act: 0=identity 1=tanh 2=relu 3=sigmoid
#define PREDS 0x53521110u
#define ACTS  0x21310223u

// ---------------------------------------------------------------------------
// Weight convert: src f32 row-major [K][1700] -> dst bf16 col-major
// [1728 cols][kpad], zero-padded in N (cols >= 1700) and K (k >= K).
// ---------------------------------------------------------------------------
__global__ __launch_bounds__(256) void convert_w(
    const float* __restrict__ src, unsigned short* __restrict__ dst_,
    int K, int kpad) {
  __bf16* dst = reinterpret_cast<__bf16*>(dst_);
  int idx = blockIdx.x * 256 + threadIdx.x;        // < 1728*kpad exactly
  int n = idx / kpad;
  int k = idx - n * kpad;
  float v = (k < K && n < 2 * NHID) ? *GCP(float, src + k * (2 * NHID) + n)
                                    : 0.0f;
  *GP(__bf16, dst + idx) = (__bf16)v;
}

// ---------------------------------------------------------------------------
// Device-wide barrier (all NBLK blocks co-resident: no LDS, 27 blocks).
// Release: per-thread fence (waits own stores to device scope) + block sync,
// then one thread arrives on an L2 atomic and spins. Acquire: block sync +
// per-thread fence (invalidates stale WGP-cache lines).
// ---------------------------------------------------------------------------
__device__ inline void grid_sync(unsigned int* bar, unsigned int target) {
  __threadfence();
  __syncthreads();
  if (threadIdx.x == 0) {
    atomicAdd(bar, 1u);
    while (*((volatile AS1 unsigned int*)(unsigned long long)bar) < target)
      __builtin_amdgcn_s_sleep(1);
  }
  __syncthreads();
  __threadfence();
}

// ---------------------------------------------------------------------------
// One GEMM+gate phase. wave w (0..215): M-block mt=w/54 (64 rows), N-tile
// nt=w%54 (16 cols of s). For the tile: c = A*W[:,n], h = A*W[:,n+850],
//   s = p + sigmoid(c) * (act(h) - p)
// A: bf16 [256][lda];  Wt: bf16 col-major [1728][ldw];  pred/s_out: f32
// [256][850];  sb: bf16 [256][864] (K-pad columns written as zero).
// ---------------------------------------------------------------------------
__device__ void gemm_phase(const __bf16* A, int lda,
                           const __bf16* Wt, int ldw,
                           const float* pred,
                           float* s_out,
                           __bf16* sb,
                           int kiters, int act) {
  int wave = blockIdx.x * 8 + (threadIdx.x >> 5);  // 0..215
  int lane = threadIdx.x & 31;
  int mt = wave / NTILES;                          // 0..3
  int nt = wave - mt * NTILES;                     // 0..53
  int m0 = mt << 6;                                // 64-row block
  int n0 = nt << 4;
  int l  = lane & 15;
  int hi = lane >> 4;

  // A 16x32 fragment layout: elems 0..7 at K=k0+hi*8, elems 8..15 at +16
  const AS1 __bf16* arow = GCP(__bf16, A + (m0 + l) * lda + hi * 8);
  // B 32x16 fragment layout: elems 0..7 at K=k0+hi*8, elems 8..15 at +16 (col-major W)
  const AS1 __bf16* bc = GCP(__bf16, Wt + (n0 + l) * ldw + hi * 8);
  const AS1 __bf16* bh = GCP(__bf16, Wt + (n0 + l + NHID) * ldw + hi * 8);

  v8f accC[4] = {};
  v8f accH[4] = {};
  for (int ki = 0; ki < kiters; ++ki) {
    int k0 = ki * 32;
    v8bf b0 = *(const AS1 v8bf*)(bc + k0);
    v8bf b1 = *(const AS1 v8bf*)(bc + k0 + 16);
    v8bf b2 = *(const AS1 v8bf*)(bh + k0);
    v8bf b3 = *(const AS1 v8bf*)(bh + k0 + 16);
    v16bf bcv, bhv;
#pragma unroll
    for (int i = 0; i < 8; ++i) {
      bcv[i] = b0[i]; bcv[i + 8] = b1[i];
      bhv[i] = b2[i]; bhv[i + 8] = b3[i];
    }
#pragma unroll
    for (int mi = 0; mi < 4; ++mi) {
      const AS1 __bf16* ar = arow + mi * 16 * lda;
      v8bf a0 = *(const AS1 v8bf*)(ar + k0);
      v8bf a1 = *(const AS1 v8bf*)(ar + k0 + 16);
      v16bf av;
#pragma unroll
      for (int i = 0; i < 8; ++i) { av[i] = a0[i]; av[i + 8] = a1[i]; }
      accC[mi] = __builtin_amdgcn_wmma_f32_16x16x32_bf16(
          false, av, false, bcv, (short)0, accC[mi], false, false);
      accH[mi] = __builtin_amdgcn_wmma_f32_16x16x32_bf16(
          false, av, false, bhv, (short)0, accH[mi], false, false);
    }
  }

  int n = n0 + l;
#pragma unroll
  for (int mi = 0; mi < 4; ++mi) {
    if (n < NHID) {
#pragma unroll
      for (int r = 0; r < 8; ++r) {
        int m = m0 + mi * 16 + hi * 8 + r;     // C/D layout: vgpr r -> row r(+8)
        float p = *GCP(float, pred + m * NHID + n);
        float c = accC[mi][r];
        float h = accH[mi][r];
        float g = 1.0f / (1.0f + __expf(-c));
        float a;
        if (act == 1)      a = tanhf(h);
        else if (act == 2) a = fmaxf(h, 0.0f);
        else if (act == 3) a = 1.0f / (1.0f + __expf(-h));
        else               a = h;
        float s = p + g * (a - p);
        *GP(float, s_out + m * NHID + n) = s;
        *GP(__bf16, sb + m * KP + n) = (__bf16)s;
      }
    } else {
#pragma unroll
      for (int r = 0; r < 8; ++r)
        *GP(__bf16, sb + (m0 + mi * 16 + hi * 8 + r) * KP + n) = (__bf16)0.0f;
    }
  }
}

// ---------------------------------------------------------------------------
// Persistent kernel: whole 128-step scan, 10 grid barriers per step.
// ---------------------------------------------------------------------------
__global__ __launch_bounds__(256) void darts_rnn(
    const float* __restrict__ inputs, const float* __restrict__ hidden,
    const unsigned short* __restrict__ W0t_,
    const unsigned short* __restrict__ Wst_,
    unsigned short* __restrict__ xh_, float* __restrict__ s_f32,
    unsigned short* __restrict__ sb_, float* __restrict__ out,
    unsigned int* __restrict__ bar) {
  const __bf16* W0t = reinterpret_cast<const __bf16*>(W0t_);
  const __bf16* Wst = reinterpret_cast<const __bf16*>(Wst_);
  __bf16* xh = reinterpret_cast<__bf16*>(xh_);
  __bf16* sb = reinterpret_cast<__bf16*>(sb_);

  int gtid = blockIdx.x * 256 + threadIdx.x;
  unsigned int target = 0;

  // init: xh = [bf16(x_0) | bf16(h_0) | 0-pad]
  for (int idx = gtid; idx < BATCH * K0P; idx += NTHREAD) {
    int m = idx / K0P;
    int j = idx - m * K0P;
    float v = (j < NHID)     ? *GCP(float, inputs + m * NHID + j)
            : (j < 2 * NHID) ? *GCP(float, hidden + m * NHID + (j - NHID))
                             : 0.0f;
    *GP(__bf16, xh + idx) = (__bf16)v;
  }
  grid_sync(bar, target += gridDim.x);

  for (int t = 0; t < T_STEPS; ++t) {
    const float* hprev = (t == 0) ? hidden : out + (size_t)(t - 1) * BH;

    // node 0: s0 = h + sigmoid(c0)*(tanh(h0) - h), K = 1728
    gemm_phase(xh, K0P, W0t, K0P, hprev, s_f32, sb, K0P / 32, 1);
    grid_sync(bar, target += gridDim.x);

    for (int i = 0; i < 8; ++i) {
      int p = (PREDS >> (4 * i)) & 15;
      int a = (ACTS >> (4 * i)) & 15;
      gemm_phase(sb + p * BATCH * KP, KP, Wst + (size_t)i * NWCOLS * KP, KP,
                 s_f32 + p * BH, s_f32 + (i + 1) * BH,
                 sb + (i + 1) * BATCH * KP, KP / 32, a);
      grid_sync(bar, target += gridDim.x);
    }

    // finish: out[t] = mean(states[1..8]); feed h back and load x[t+1]
    float* ot = out + (size_t)t * BH;
    const float* xnext = inputs + (size_t)(t + 1) * BH;
    for (int idx = gtid; idx < BH; idx += NTHREAD) {
      int m = idx / NHID;
      int nn = idx - m * NHID;
      float sum = 0.0f;
#pragma unroll
      for (int i = 1; i <= 8; ++i) sum += *GCP(float, s_f32 + i * BH + idx);
      float v = sum * 0.125f;
      *GP(float, ot + idx) = v;
      *GP(__bf16, xh + m * K0P + NHID + nn) = (__bf16)v;
      if (t + 1 < T_STEPS)
        *GP(__bf16, xh + m * K0P + nn) = (__bf16)*GCP(float, xnext + idx);
    }
    grid_sync(bar, target += gridDim.x);
  }
}

// ---------------------------------------------------------------------------
extern "C" void kernel_launch(void* const* d_in, const int* in_sizes, int n_in,
                              void* d_out, int out_size, void* d_ws, size_t ws_size,
                              hipStream_t stream) {
  (void)in_sizes; (void)n_in; (void)out_size; (void)ws_size;
  const float* inputs = (const float*)d_in[0];   // [128,256,850]
  const float* hidden = (const float*)d_in[1];   // [1,256,850]
  const float* W0     = (const float*)d_in[2];   // [1700,1700]
  const float* Ws     = (const float*)d_in[3];   // [8,850,1700]
  float* out = (float*)d_out;                    // [128,256,850] ++ [1,256,850]

  // workspace layout (all 16B aligned)
  unsigned short* W0t   = (unsigned short*)d_ws;              // 1728*1728 bf16
  unsigned short* Wst   = W0t + (size_t)NWCOLS * K0P;         // 8*1728*864 bf16
  unsigned short* xh    = Wst + (size_t)8 * NWCOLS * KP;      // 256*1728 bf16
  float*          s_f32 = (float*)(xh + (size_t)BATCH * K0P); // 9*256*850 f32
  unsigned short* s_b16 = (unsigned short*)(s_f32 + (size_t)9 * BH); // 9*256*864
  unsigned int*   bar   = (unsigned int*)(s_b16 + (size_t)9 * BATCH * KP);

  // one-off weight conversion (row-major f32 -> col-major bf16, padded)
  convert_w<<<(NWCOLS * K0P) / 256, 256, 0, stream>>>(W0, W0t, K0, K0P);
  for (int i = 0; i < 8; ++i)
    convert_w<<<(NWCOLS * KP) / 256, 256, 0, stream>>>(
        Ws + (size_t)i * NHID * (2 * NHID), Wst + (size_t)i * NWCOLS * KP,
        NHID, KP);

  (void)hipMemsetAsync(bar, 0, sizeof(unsigned int), stream);

  darts_rnn<<<NBLK, 256, 0, stream>>>(inputs, hidden, W0t, Wst, xh, s_f32,
                                      s_b16, out, bar);

  // second tuple output: hiddens[-1][None]
  (void)hipMemcpyAsync(out + (size_t)T_STEPS * BH,
                       out + (size_t)(T_STEPS - 1) * BH,
                       (size_t)BH * sizeof(float), hipMemcpyDeviceToDevice,
                       stream);
}
